// ScatterOPTBlock_687194768149
// MI455X (gfx1250) — matrix operand order repaired and checked by reading, not compiled, not run on hardware
//
#include <hip/hip_runtime.h>
#include <hip/hip_bf16.h>

typedef __attribute__((ext_vector_type(16))) _Float16 v16h;
typedef __attribute__((ext_vector_type(8)))  float    v8f;

#define NPTS 262144
#define CDIM 128
#define BBAT 8
#define HID  256
#define TWIN 128
#define MAXB (NPTS / TWIN + BBAT)   // 2056
#define ATT_SCALE 0.17677669529663687f   // (128/4)^-0.5
#define EPS_LN 1e-5f

union V16 { v16h v; uint4 q[2]; };

enum { EPI_STOREH = 0, EPI_SEG = 1, EPI_GATE = 2, EPI_PROJ = 3 };

// ---------------------------------------------------------------------------
// zero a float buffer
__global__ void zero_kernel(float* __restrict__ p, int n) {
    int i = blockIdx.x * blockDim.x + threadIdx.x;
    if (i < n) p[i] = 0.0f;
}

// ---------------------------------------------------------------------------
// cast + transpose weights f32[K][Cout] -> f16[Cout][K]
__global__ void wcast_kernel(const float* __restrict__ w, _Float16* __restrict__ wt,
                             int K, int Cout) {
    int i = blockIdx.x * blockDim.x + threadIdx.x;
    if (i >= K * Cout) return;
    int k = i / Cout, c = i - k * Cout;       // coalesced read over c
    wt[(size_t)c * K + k] = (_Float16)w[i];
}

// ---------------------------------------------------------------------------
// bucket id per point + bucket counts
__global__ void bucket_kernel(const int* __restrict__ batch, const int* __restrict__ lengths,
                              const float* __restrict__ scores, int* __restrict__ gb,
                              float* __restrict__ cnt) {
    int i = blockIdx.x * blockDim.x + threadIdx.x;
    if (i >= NPTS) return;
    int bb = batch[i];
    int off = 0, nb = 1;
#pragma unroll
    for (int j = 0; j < BBAT; ++j) {
        int nbj = (lengths[j] + TWIN - 1) / TWIN;
        if (j < bb)  off += nbj;
        if (j == bb) nb = nbj;
    }
    int loc = (int)(scores[i] * (float)nb);
    if (loc > nb - 1) loc = nb - 1;
    int g = off + loc;
    gb[i] = g;
    atomicAdd(&cnt[g], 1.0f);
}

// ---------------------------------------------------------------------------
// row LayerNorm (C=128) + cast to f16.  One wave per row, lane holds 4 cols.
__global__ __launch_bounds__(256)
void ln_cast_kernel(const float* __restrict__ x, const float* __restrict__ g,
                    const float* __restrict__ b, _Float16* __restrict__ out) {
    int wave = threadIdx.x >> 5, lane = threadIdx.x & 31;
    int row = blockIdx.x * 8 + wave;
    if (row >= NPTS) return;
    const float4 v = ((const float4*)(x + (size_t)row * CDIM))[lane];
    float s  = v.x + v.y + v.z + v.w;
    float s2 = v.x * v.x + v.y * v.y + v.z * v.z + v.w * v.w;
#pragma unroll
    for (int m = 16; m > 0; m >>= 1) {
        s  += __shfl_xor(s,  m, 32);
        s2 += __shfl_xor(s2, m, 32);
    }
    float mu  = s * (1.0f / CDIM);
    float var = s2 * (1.0f / CDIM) - mu * mu;
    float rs  = rsqrtf(var + EPS_LN);
    int c = lane * 4;
    union { _Float16 h[4]; uint2 u; } oo;
    oo.h[0] = (_Float16)((v.x - mu) * rs * g[c + 0] + b[c + 0]);
    oo.h[1] = (_Float16)((v.y - mu) * rs * g[c + 1] + b[c + 1]);
    oo.h[2] = (_Float16)((v.z - mu) * rs * g[c + 2] + b[c + 2]);
    oo.h[3] = (_Float16)((v.w - mu) * rs * g[c + 3] + b[c + 3]);
    ((uint2*)(out + (size_t)row * CDIM))[lane] = oo.u;
}

// ---------------------------------------------------------------------------
// th = q * (ksum[gb]/cnt) * SCALE     (f16 out)
__global__ void gatein_kernel(const _Float16* __restrict__ qh, const int* __restrict__ gb,
                              const float* __restrict__ ksum, const float* __restrict__ cnt,
                              _Float16* __restrict__ th) {
    int i = blockIdx.x * blockDim.x + threadIdx.x;
    if (i >= NPTS * CDIM) return;
    int row = i >> 7, col = i & 127;
    int g = gb[row];
    float c = fmaxf(cnt[g], 1.0f);
    th[i] = (_Float16)((float)qh[i] * (ksum[g * CDIM + col] / c) * ATT_SCALE);
}

// ---------------------------------------------------------------------------
// Generic K=128, Cout=128 WMMA GEMM: out = A[128-row tile] @ Wt^T + bias, fused epilogues.
// Wt stored transposed [Cout][K] f16.  Block = 256 threads = 8 waves; wave w owns
// rows [16w,16w+16) x all 8 col tiles.
__global__ __launch_bounds__(256)
void gemm128_kernel(const _Float16* __restrict__ A, const _Float16* __restrict__ Wt,
                    const float* __restrict__ bias, int epi,
                    _Float16* __restrict__ outH, float* __restrict__ outF,
                    const int* __restrict__ gb, float* __restrict__ seg,
                    const float* __restrict__ vsum, const float* __restrict__ cnt,
                    const float* __restrict__ feat) {
    __shared__ _Float16 As[128 * 136];
    const int tid = threadIdx.x;
    const int rowBase = blockIdx.x * 128;

    // stage A tile: 128 rows x 16 chunks of 8 halves
    for (int c8 = tid; c8 < 128 * 16; c8 += 256) {
        int r = c8 >> 4, cc = c8 & 15;
        uint4 d = ((const uint4*)(A + (size_t)(rowBase + r) * CDIM))[cc];
        *(uint4*)(&As[r * 136 + cc * 8]) = d;
    }
    __syncthreads();

    const int wave = tid >> 5, lane = tid & 31;
    const int r0 = lane & 15, hi = lane >> 4;

    v8f acc[8] = {};
#pragma unroll
    for (int ks = 0; ks < 4; ++ks) {
        V16 a;
        const _Float16* ap = &As[(wave * 16 + r0) * 136 + ks * 32 + hi * 8];
        a.q[0] = *(const uint4*)ap;
        a.q[1] = *(const uint4*)(ap + 16);
#pragma unroll
        for (int ct = 0; ct < 8; ++ct) {
            V16 bfr;
            const _Float16* bp = Wt + (size_t)(ct * 16 + r0) * 128 + ks * 32 + hi * 8;
            bfr.q[0] = *(const uint4*)bp;
            bfr.q[1] = *(const uint4*)(bp + 16);
            acc[ct] = __builtin_amdgcn_wmma_f32_16x16x32_f16(
                false, a.v, false, bfr.v, (short)0, acc[ct], false, false);
        }
    }

#pragma unroll
    for (int ct = 0; ct < 8; ++ct) {
        int col = ct * 16 + r0;
        float bv = bias[col];
#pragma unroll
        for (int r2 = 0; r2 < 8; ++r2) {
            int row = rowBase + wave * 16 + r2 + hi * 8;
            float val = acc[ct][r2] + bv;
            if (epi == EPI_STOREH) {
                outH[(size_t)row * 128 + col] = (_Float16)val;
            } else if (epi == EPI_SEG) {
                atomicAdd(&seg[gb[row] * 128 + col], val);
            } else if (epi == EPI_GATE) {
                int g = gb[row];
                float cn = fmaxf(cnt[g], 1.0f);
                float gate = 1.0f / (1.0f + __expf(-val));
                outH[(size_t)row * 128 + col] = (_Float16)(gate * (vsum[g * 128 + col] / cn));
            } else { // EPI_PROJ: x = feat + attn_out
                outF[(size_t)row * 128 + col] = feat[(size_t)row * 128 + col] + val;
            }
        }
    }
}

// ---------------------------------------------------------------------------
// FFN1: [128 rows] x (K=128 -> N=256), fused LayerNorm(gf,bf) + ReLU, f16 out.
// Each wave owns a full 16-row stripe across all 256 cols -> in-register row LN.
__global__ __launch_bounds__(256)
void ffn1_kernel(const _Float16* __restrict__ A, const _Float16* __restrict__ Wt,
                 const float* __restrict__ bias, const float* __restrict__ gf,
                 const float* __restrict__ bfv, _Float16* __restrict__ outH) {
    __shared__ _Float16 As[128 * 136];
    const int tid = threadIdx.x;
    const int rowBase = blockIdx.x * 128;

    for (int c8 = tid; c8 < 128 * 16; c8 += 256) {
        int r = c8 >> 4, cc = c8 & 15;
        uint4 d = ((const uint4*)(A + (size_t)(rowBase + r) * CDIM))[cc];
        *(uint4*)(&As[r * 136 + cc * 8]) = d;
    }
    __syncthreads();

    const int wave = tid >> 5, lane = tid & 31;
    const int r0 = lane & 15, hi = lane >> 4;

    v8f acc[16] = {};
#pragma unroll
    for (int ks = 0; ks < 4; ++ks) {
        V16 a;
        const _Float16* ap = &As[(wave * 16 + r0) * 136 + ks * 32 + hi * 8];
        a.q[0] = *(const uint4*)ap;
        a.q[1] = *(const uint4*)(ap + 16);
#pragma unroll
        for (int ct = 0; ct < 16; ++ct) {
            V16 bfr;
            const _Float16* bp = Wt + (size_t)(ct * 16 + r0) * 128 + ks * 32 + hi * 8;
            bfr.q[0] = *(const uint4*)bp;
            bfr.q[1] = *(const uint4*)(bp + 16);
            acc[ct] = __builtin_amdgcn_wmma_f32_16x16x32_f16(
                false, a.v, false, bfr.v, (short)0, acc[ct], false, false);
        }
    }

    // bias + per-row stats (rows r2+8*hi; 16 lanes of a half share the rows)
    float s[8] = {}, s2[8] = {};
#pragma unroll
    for (int ct = 0; ct < 16; ++ct) {
        float bv = bias[ct * 16 + r0];
#pragma unroll
        for (int r2 = 0; r2 < 8; ++r2) {
            float val = acc[ct][r2] + bv;
            acc[ct][r2] = val;
            s[r2]  += val;
            s2[r2] += val * val;
        }
    }
#pragma unroll
    for (int r2 = 0; r2 < 8; ++r2) {
        float ss = s[r2], qq = s2[r2];
#pragma unroll
        for (int m = 1; m < 16; m <<= 1) {   // reduce within 16-lane half
            ss += __shfl_xor(ss, m, 32);
            qq += __shfl_xor(qq, m, 32);
        }
        float mu  = ss * (1.0f / HID);
        float var = qq * (1.0f / HID) - mu * mu;
        s[r2]  = mu;
        s2[r2] = rsqrtf(var + EPS_LN);
    }
#pragma unroll
    for (int ct = 0; ct < 16; ++ct) {
        int col = ct * 16 + r0;
        float gv = gf[col], bb = bfv[col];
#pragma unroll
        for (int r2 = 0; r2 < 8; ++r2) {
            int row = rowBase + wave * 16 + r2 + hi * 8;
            float v = (acc[ct][r2] - s[r2]) * s2[r2] * gv + bb;
            v = fmaxf(v, 0.0f);
            outH[(size_t)row * HID + col] = (_Float16)v;
        }
    }
}

// ---------------------------------------------------------------------------
// FFN2: [64 rows] x (K=256 -> N=128), epilogue: d_out += acc + bias.
__global__ __launch_bounds__(256)
void ffn2_kernel(const _Float16* __restrict__ A, const _Float16* __restrict__ Wt,
                 const float* __restrict__ bias, float* __restrict__ outF) {
    __shared__ _Float16 As[64 * 264];
    const int tid = threadIdx.x;
    const int rowBase = blockIdx.x * 64;

    for (int c8 = tid; c8 < 64 * 32; c8 += 256) {
        int r = c8 >> 5, cc = c8 & 31;
        uint4 d = ((const uint4*)(A + (size_t)(rowBase + r) * HID))[cc];
        *(uint4*)(&As[r * 264 + cc * 8]) = d;
    }
    __syncthreads();

    const int wave = tid >> 5, lane = tid & 31;
    const int r0 = lane & 15, hi = lane >> 4;
    const int stripe = wave >> 1;           // 4 stripes of 16 rows
    const int chalf  = wave & 1;            // each wave: 4 of the 8 col tiles

    v8f acc[4] = {};
#pragma unroll
    for (int ks = 0; ks < 8; ++ks) {
        V16 a;
        const _Float16* ap = &As[(stripe * 16 + r0) * 264 + ks * 32 + hi * 8];
        a.q[0] = *(const uint4*)ap;
        a.q[1] = *(const uint4*)(ap + 16);
#pragma unroll
        for (int ct = 0; ct < 4; ++ct) {
            int ctg = chalf * 4 + ct;
            V16 bfr;
            const _Float16* bp = Wt + (size_t)(ctg * 16 + r0) * HID + ks * 32 + hi * 8;
            bfr.q[0] = *(const uint4*)bp;
            bfr.q[1] = *(const uint4*)(bp + 16);
            acc[ct] = __builtin_amdgcn_wmma_f32_16x16x32_f16(
                false, a.v, false, bfr.v, (short)0, acc[ct], false, false);
        }
    }

#pragma unroll
    for (int ct = 0; ct < 4; ++ct) {
        int col = (chalf * 4 + ct) * 16 + r0;
        float bv = bias[col];
#pragma unroll
        for (int r2 = 0; r2 < 8; ++r2) {
            int row = rowBase + stripe * 16 + r2 + hi * 8;
            size_t idx = (size_t)row * 128 + col;
            outF[idx] = outF[idx] + acc[ct][r2] + bv;
        }
    }
}

// ---------------------------------------------------------------------------
extern "C" void kernel_launch(void* const* d_in, const int* in_sizes, int n_in,
                              void* d_out, int out_size, void* d_ws, size_t ws_size,
                              hipStream_t stream) {
    const float* feat    = (const float*)d_in[0];
    const int*   batch   = (const int*)d_in[1];
    const int*   lengths = (const int*)d_in[2];
    const float* scores  = (const float*)d_in[3];
    const float* Wq = (const float*)d_in[4];  const float* bq = (const float*)d_in[5];
    const float* Wk = (const float*)d_in[6];  const float* bk = (const float*)d_in[7];
    const float* Wv = (const float*)d_in[8];  const float* bv = (const float*)d_in[9];
    const float* Wg = (const float*)d_in[10]; const float* bg = (const float*)d_in[11];
    const float* Wp = (const float*)d_in[12]; const float* bp = (const float*)d_in[13];
    const float* g1 = (const float*)d_in[14]; const float* b1 = (const float*)d_in[15];
    const float* g2 = (const float*)d_in[16]; const float* b2 = (const float*)d_in[17];
    const float* Wf1 = (const float*)d_in[18]; const float* bf1 = (const float*)d_in[19];
    const float* gf  = (const float*)d_in[20]; const float* bfv = (const float*)d_in[21];
    const float* Wf2 = (const float*)d_in[22]; const float* bf2 = (const float*)d_in[23];
    float* out = (float*)d_out;

    // ---- workspace carve-up ----
    char* ws = (char*)d_ws;
    size_t off = 0;
    auto carve = [&](size_t bytes) { size_t o = off; off = (off + bytes + 255) & ~(size_t)255; return o; };
    int*      gb    = (int*)(ws + carve((size_t)NPTS * 4));
    float*    ksum  = (float*)(ws + carve(((size_t)2 * MAXB * CDIM + MAXB) * 4));
    float*    vsum  = ksum + (size_t)MAXB * CDIM;
    float*    cnt   = vsum + (size_t)MAXB * CDIM;
    _Float16* WqT   = (_Float16*)(ws + carve((size_t)(5 * 128 * 128 + 2 * 128 * 256) * 2));
    _Float16* WkT   = WqT + 128 * 128;
    _Float16* WvT   = WkT + 128 * 128;
    _Float16* WgT   = WvT + 128 * 128;
    _Float16* WpT   = WgT + 128 * 128;
    _Float16* Wf1T  = WpT + 128 * 128;
    _Float16* Wf2T  = Wf1T + 128 * 256;
    _Float16* xh    = (_Float16*)(ws + carve((size_t)NPTS * CDIM * 2));  // xn / th / hh
    _Float16* qh    = (_Float16*)(ws + carve((size_t)NPTS * CDIM * 2));  // q / u
    _Float16* fh    = (_Float16*)(ws + carve((size_t)NPTS * HID * 2));   // ffn hidden

    // ---- 0: zero segment accumulators (must be re-done every call) ----
    int nz = 2 * MAXB * CDIM + MAXB;
    zero_kernel<<<(nz + 255) / 256, 256, 0, stream>>>(ksum, nz);

    // ---- 1: weight cast + transpose ----
    wcast_kernel<<<(128 * 128 + 255) / 256, 256, 0, stream>>>(Wq, WqT, 128, 128);
    wcast_kernel<<<(128 * 128 + 255) / 256, 256, 0, stream>>>(Wk, WkT, 128, 128);
    wcast_kernel<<<(128 * 128 + 255) / 256, 256, 0, stream>>>(Wv, WvT, 128, 128);
    wcast_kernel<<<(128 * 128 + 255) / 256, 256, 0, stream>>>(Wg, WgT, 128, 128);
    wcast_kernel<<<(128 * 128 + 255) / 256, 256, 0, stream>>>(Wp, WpT, 128, 128);
    wcast_kernel<<<(128 * 256 + 255) / 256, 256, 0, stream>>>(Wf1, Wf1T, 128, 256);
    wcast_kernel<<<(256 * 128 + 255) / 256, 256, 0, stream>>>(Wf2, Wf2T, 256, 128);

    // ---- 2: bucket ids + counts ----
    bucket_kernel<<<NPTS / 256, 256, 0, stream>>>(batch, lengths, scores, gb, cnt);

    // ---- 3: x = LN(feat) -> f16 ----
    ln_cast_kernel<<<NPTS / 8, 256, 0, stream>>>(feat, g1, b1, xh);

    // ---- 4: q/k/v GEMMs (k,v fused into segment sums) ----
    gemm128_kernel<<<NPTS / 128, 256, 0, stream>>>(xh, WqT, bq, EPI_STOREH,
        qh, nullptr, gb, nullptr, nullptr, nullptr, nullptr);
    gemm128_kernel<<<NPTS / 128, 256, 0, stream>>>(xh, WkT, bk, EPI_SEG,
        nullptr, nullptr, gb, ksum, nullptr, nullptr, nullptr);
    gemm128_kernel<<<NPTS / 128, 256, 0, stream>>>(xh, WvT, bv, EPI_SEG,
        nullptr, nullptr, gb, vsum, nullptr, nullptr, nullptr);

    // ---- 5: th = q * k_ctx * SCALE ----
    gatein_kernel<<<(NPTS * CDIM) / 256, 256, 0, stream>>>(qh, gb, ksum, cnt, xh);

    // ---- 6: gate GEMM -> sigmoid -> * v_ctx ----
    gemm128_kernel<<<NPTS / 128, 256, 0, stream>>>(xh, WgT, bg, EPI_GATE,
        qh, nullptr, gb, nullptr, vsum, cnt, nullptr);

    // ---- 7: proj GEMM + residual -> d_out (f32) ----
    gemm128_kernel<<<NPTS / 128, 256, 0, stream>>>(qh, WpT, bp, EPI_PROJ,
        nullptr, out, gb, nullptr, nullptr, nullptr, feat);

    // ---- 8: h = LN(x) -> f16 ----
    ln_cast_kernel<<<NPTS / 8, 256, 0, stream>>>(out, g2, b2, xh);

    // ---- 9: FFN1 + LN + ReLU ----
    ffn1_kernel<<<NPTS / 128, 256, 0, stream>>>(xh, Wf1T, bf1, gf, bfv, fh);

    // ---- 10: FFN2, accumulate into d_out ----
    ffn2_kernel<<<NPTS / 64, 256, 0, stream>>>(fh, Wf2T, bf2, out);
}
